// RNN_16587163697185
// MI455X (gfx1250) — compile-verified, matrix operand order checked by reading, not run on hardware
//
// MI455X (gfx1250) 3-layer LSTM + dense head.
// bf16 WMMA (16x16x32, f32 accum) for all GEMMs; f32 pre-activations & cell state.
// GEMM: double-buffered LDS tiles, A-tile staged with GLOBAL_LOAD_ASYNC_TO_LDS_B128
// (ASYNCcnt DMA, overlapped with WMMA), one barrier per K-slab.
// Scan: 64 persistent WGs, Wh slice resident in LDS (128KB) across all 256 steps,
// device-wide atomic barrier per step, double-buffered h in global (L2-hot).

#include <hip/hip_runtime.h>
#include <stdint.h>

#define B_    32
#define T_    256
#define H_    1024
#define V_    65
#define FOURH 4096
#define NROWS (B_ * T_)   // 8192
#define NWG_SCAN 64

typedef __bf16        v16bf __attribute__((ext_vector_type(16)));
typedef float         v8f   __attribute__((ext_vector_type(8)));
typedef unsigned int  u32x4 __attribute__((ext_vector_type(4)));

union FragBF {
    v16bf v;
    u32x4 q[2];
    unsigned short s[16];
};
union AccF {
    v8f   v;
    float f[8];
};

__device__ __forceinline__ unsigned short f2bf(float f) {
    unsigned int u = __float_as_uint(f);
    u += 0x7FFFu + ((u >> 16) & 1u);           // round-to-nearest-even
    return (unsigned short)(u >> 16);
}
__device__ __forceinline__ float sig_(float x)  { return 1.0f / (1.0f + __expf(-x)); }
__device__ __forceinline__ float tanh_(float x) { return 2.0f / (1.0f + __expf(-2.0f * x)) - 1.0f; }

// Generic pointers to LDS are {SHARED_BASE[63:32], lds_offset[31:0]} (ISA 10.2),
// so the low 32 bits of a __shared__ pointer are the LDS byte address.
__device__ __forceinline__ unsigned lds_off(const void* p) {
    return (unsigned)(uintptr_t)p;
}
// Async global->LDS 16B copy (VGLOBAL SADDR form: VDST=LDS addr, VADDR=u32 offset).
__device__ __forceinline__ void async_copy_b128(unsigned lds_dst, unsigned goff,
                                                const void* base) {
    asm volatile("global_load_async_to_lds_b128 %0, %1, %2"
                 :: "v"(lds_dst), "v"(goff),
                    "s"((unsigned long long)(uintptr_t)base)
                 : "memory");
}
__device__ __forceinline__ void async_wait0() {
    asm volatile("s_wait_asynccnt 0x0" ::: "memory");
}

// ---------------------------------------------------------------- conversions
__global__ __launch_bounds__(256) void f32_to_bf16_k(const float* __restrict__ src,
                                                     unsigned short* __restrict__ dst,
                                                     long n) {
    long i = (long)blockIdx.x * 256 + threadIdx.x;
    long stride = (long)gridDim.x * 256;
    for (; i < n; i += stride) dst[i] = f2bf(src[i]);
}

__global__ void init_barrier_k(int* bar) {
    if (threadIdx.x == 0) { bar[0] = 0; bar[1] = 0; }
}

// ---------------------------------------------------------------- device barrier
__device__ __forceinline__ void grid_barrier(int* bar) {
    __syncthreads();
    if (threadIdx.x == 0) {
        int g = __hip_atomic_load(&bar[1], __ATOMIC_ACQUIRE, __HIP_MEMORY_SCOPE_AGENT);
        int old = __hip_atomic_fetch_add(&bar[0], 1, __ATOMIC_ACQ_REL, __HIP_MEMORY_SCOPE_AGENT);
        if (old == NWG_SCAN - 1) {
            // reset arrival count BEFORE releasing the generation
            __hip_atomic_store(&bar[0], 0, __ATOMIC_RELAXED, __HIP_MEMORY_SCOPE_AGENT);
            __hip_atomic_fetch_add(&bar[1], 1, __ATOMIC_RELEASE, __HIP_MEMORY_SCOPE_AGENT);
        } else {
            while (__hip_atomic_load(&bar[1], __ATOMIC_ACQUIRE, __HIP_MEMORY_SCOPE_AGENT) == g) {
                __builtin_amdgcn_s_sleep(2);
            }
        }
    }
    __syncthreads();
}

// ---------------------------------------------------------------- big GEMM
// C[M][N] = A[M][K](bf16,row-major) * B[K][N](bf16,row-major) + bias[N]  (f32 out)
// block: 256 threads (8 waves), 128x128 tile; wave = 32x64 (2x4 wmma accums)
// Double-buffered LDS; A-tile via async-to-LDS DMA, B-tile transposed via VGPRs.
__device__ __forceinline__ void stage_tiles(const unsigned short* __restrict__ A,
                                            const unsigned short* __restrict__ Bm,
                                            int blockM0, int blockN0,
                                            int M, int N, int K, int k0,
                                            unsigned short (*sAbuf)[32],
                                            unsigned short (*sBbuf)[32],
                                            int tid)
{
    // A tile: 128 rows x 32 k as 16B async DMA chunks (512 chunks / 256 thr)
    for (int e = tid; e < 128 * 4; e += 256) {
        int r = e >> 2, c = (e & 3) * 8;
        int gr = blockM0 + r;
        if (gr < M) {
            async_copy_b128(lds_off(&sAbuf[r][c]),
                            (unsigned)(((size_t)gr * K + k0 + c) * 2), A);
        } else {
            u32x4 z = {0u, 0u, 0u, 0u};
            *reinterpret_cast<u32x4*>(&sAbuf[r][c]) = z;
        }
    }
    // B tile: 32 k x 128 cols, stored transposed (col-major, K-contiguous) in LDS
    for (int e = tid; e < 32 * 16; e += 256) {
        int kk = e >> 4, c8 = (e & 15) * 8;
        int gc = blockN0 + c8;
        FragBF w;
#pragma unroll
        for (int i = 0; i < 16; i++) w.s[i] = 0;
        if (((N & 7) == 0) && (gc + 7 < N)) {
            w.q[0] = *reinterpret_cast<const u32x4*>(&Bm[(size_t)(k0 + kk) * N + gc]);
        } else {
            for (int i = 0; i < 8; i++)
                if (gc + i < N) w.s[i] = Bm[(size_t)(k0 + kk) * N + gc + i];
        }
#pragma unroll
        for (int i = 0; i < 8; i++) sBbuf[c8 + i][kk] = w.s[i];
    }
}

__global__ __launch_bounds__(256)
void gemm_bf16_bias_k(const unsigned short* __restrict__ A,
                      const unsigned short* __restrict__ B,
                      const float* __restrict__ bias,
                      float* __restrict__ C,
                      int M, int N, int K)
{
    __shared__ __align__(16) unsigned short sA[2][128][32];   // [buf][row][k]
    __shared__ __align__(16) unsigned short sB[2][128][32];   // [buf][col][k]

    const int tid  = threadIdx.x;
    const int lane = tid & 31;
    const int wave = tid >> 5;
    const int l15  = lane & 15;
    const int hi   = lane >> 4;
    const int waveM = wave & 3;          // 4 groups of 32 rows
    const int waveN = wave >> 2;         // 2 groups of 64 cols
    const int blockN0 = blockIdx.x * 128;
    const int blockM0 = blockIdx.y * 128;

    AccF acc[2][4];
#pragma unroll
    for (int i = 0; i < 2; i++)
#pragma unroll
        for (int j = 0; j < 4; j++)
#pragma unroll
            for (int r = 0; r < 8; r++) acc[i][j].f[r] = 0.0f;

    // prologue: stage slab 0
    stage_tiles(A, B, blockM0, blockN0, M, N, K, 0, sA[0], sB[0], tid);
    async_wait0();
    __syncthreads();

    int buf = 0;
    for (int k0 = 0; k0 < K; k0 += 32) {
        // kick off DMA + stage for next slab while computing this one
        if (k0 + 32 < K)
            stage_tiles(A, B, blockM0, blockN0, M, N, K, k0 + 32,
                        sA[buf ^ 1], sB[buf ^ 1], tid);

        unsigned short (*sAc)[32] = sA[buf];
        unsigned short (*sBc)[32] = sB[buf];

        FragBF bf[4];
#pragma unroll
        for (int nt = 0; nt < 4; nt++) {
            int col = waveN * 64 + nt * 16 + l15;
            bf[nt].q[0] = *reinterpret_cast<const u32x4*>(&sBc[col][hi * 8]);
            bf[nt].q[1] = *reinterpret_cast<const u32x4*>(&sBc[col][16 + hi * 8]);
        }
#pragma unroll
        for (int mt = 0; mt < 2; mt++) {
            FragBF af;
            int row = waveM * 32 + mt * 16 + l15;
            af.q[0] = *reinterpret_cast<const u32x4*>(&sAc[row][hi * 8]);
            af.q[1] = *reinterpret_cast<const u32x4*>(&sAc[row][16 + hi * 8]);
#pragma unroll
            for (int nt = 0; nt < 4; nt++) {
                acc[mt][nt].v = __builtin_amdgcn_wmma_f32_16x16x32_bf16(
                    false, af.v, false, bf[nt].v, (short)0, acc[mt][nt].v, false, false);
            }
        }

        async_wait0();       // own async copies landed (others covered by barrier)
        __syncthreads();     // compiler adds s_wait_dscnt for the B-tile ds_stores
        buf ^= 1;
    }

    // epilogue: C layout per ISA (VGPR r: lanes0-15 M=r, lanes16-31 M=8+r; N=lane&15)
#pragma unroll
    for (int mt = 0; mt < 2; mt++) {
#pragma unroll
        for (int nt = 0; nt < 4; nt++) {
            int col = blockN0 + waveN * 64 + nt * 16 + l15;
            if (col < N) {
                float bv = bias ? bias[col] : 0.0f;
#pragma unroll
                for (int r = 0; r < 8; r++) {
                    int row = blockM0 + waveM * 32 + mt * 16 + hi * 8 + r;
                    if (row < M) C[(size_t)row * N + col] = acc[mt][nt].f[r] + bv;
                }
            }
        }
    }
}

// ---------------------------------------------------------------- LSTM scan
// 64 WGs x 256 threads. WG g owns hidden units j in [g*16, g*16+16).
// Per step: z = xW[:,t,:] + h @ Wh  (slice: 4 gates x 16 cols), gates, update c/h.
// Wh slice lives in LDS for all 256 steps. h ping-pongs in global (L2-hot).
__global__ __launch_bounds__(256)
void lstm_scan_k(const float* __restrict__ xW,            // [8192][4096] (bias included)
                 const unsigned short* __restrict__ Wh,   // [1024][4096] bf16
                 const float* __restrict__ c0,            // [32][1024]
                 const float* __restrict__ h0,            // [32][1024]
                 unsigned short* __restrict__ o,          // [8192][1024] bf16
                 unsigned short* __restrict__ h_buf,      // 2 x [32][1024] bf16
                 int* __restrict__ bar)
{
    __shared__ __align__(16) unsigned short sWhT[4][16][1024];  // [gate][col][k] 128KB
    __shared__ __align__(16) float sZ[4][32][16];               // [gate][b][col]
    __shared__ __align__(16) float sC[32][16];                  // cell state slice

    const int tid   = threadIdx.x;
    const int wg    = blockIdx.x;         // 0..63
    const int jbase = wg * 16;
    const int lane  = tid & 31;
    const int wave  = tid >> 5;
    const int l15   = lane & 15;
    const int hi    = lane >> 4;
    const int mt    = wave & 1;           // batch half (rows 0-15 / 16-31)
    const int gate  = wave >> 1;          // 0=i 1=f 2=g 3=o

    // --- preload Wh slice into LDS, transposed to [gate][col][k] ---
    for (int e = tid; e < 4 * 1024; e += 256) {
        int g = e >> 10, k = e & 1023;
        FragBF w;
        const unsigned short* src = &Wh[(size_t)k * FOURH + g * H_ + jbase];
        w.q[0] = *reinterpret_cast<const u32x4*>(src);
        w.q[1] = *reinterpret_cast<const u32x4*>(src + 8);
#pragma unroll
        for (int c = 0; c < 16; c++) sWhT[g][c][k] = w.s[c];
    }
    // --- init cell slice (LDS) + h buffer 0 (global, bf16) ---
    for (int e = tid; e < 32 * 16; e += 256) {
        int b = e >> 4, jj = e & 15;
        sC[b][jj] = c0[b * H_ + jbase + jj];
        h_buf[b * H_ + jbase + jj] = f2bf(h0[b * H_ + jbase + jj]);
    }
    __threadfence();
    grid_barrier(bar);

    for (int t = 0; t < T_; t++) {
        const unsigned short* hr = h_buf + (t & 1) * (B_ * H_);        // read buf
        unsigned short*       hw = h_buf + ((t + 1) & 1) * (B_ * H_);  // write buf

        // --- z slice GEMM: acc[16x16] = h[mt-half] @ WhT[gate] ---
        AccF acc;
#pragma unroll
        for (int r = 0; r < 8; r++) acc.f[r] = 0.0f;

        const int arow = mt * 16 + l15;
#pragma unroll 4
        for (int ks = 0; ks < 32; ks++) {
            FragBF af, bf;
            const unsigned short* ap = &hr[arow * H_ + ks * 32 + hi * 8];
            af.q[0] = *reinterpret_cast<const u32x4*>(ap);
            af.q[1] = *reinterpret_cast<const u32x4*>(ap + 16);
            const unsigned short* bp = &sWhT[gate][l15][ks * 32 + hi * 8];
            bf.q[0] = *reinterpret_cast<const u32x4*>(bp);
            bf.q[1] = *reinterpret_cast<const u32x4*>(bp + 16);
            acc.v = __builtin_amdgcn_wmma_f32_16x16x32_bf16(
                false, af.v, false, bf.v, (short)0, acc.v, false, false);
        }
#pragma unroll
        for (int r = 0; r < 8; r++)
            sZ[gate][mt * 16 + hi * 8 + r][l15] = acc.f[r];
        __syncthreads();

        // --- gates + state update (512 elems = 32 batch x 16 j) ---
        for (int e = tid; e < B_ * 16; e += 256) {
            int b = e >> 4, jj = e & 15;
            size_t xrow = ((size_t)b * T_ + t) * FOURH;
            float zi = sZ[0][b][jj] + xW[xrow + 0 * H_ + jbase + jj];
            float zf = sZ[1][b][jj] + xW[xrow + 1 * H_ + jbase + jj];
            float zg = sZ[2][b][jj] + xW[xrow + 2 * H_ + jbase + jj];
            float zo = sZ[3][b][jj] + xW[xrow + 3 * H_ + jbase + jj];
            float cv = sig_(zf) * sC[b][jj] + sig_(zi) * tanh_(zg);
            sC[b][jj] = cv;
            float hv = sig_(zo) * tanh_(cv);
            unsigned short hb = f2bf(hv);
            hw[b * H_ + jbase + jj] = hb;
            o[((size_t)b * T_ + t) * H_ + jbase + jj] = hb;
        }
        __threadfence();
        grid_barrier(bar);
    }
}

// ---------------------------------------------------------------- host launch
extern "C" void kernel_launch(void* const* d_in, const int* in_sizes, int n_in,
                              void* d_out, int out_size, void* d_ws, size_t ws_size,
                              hipStream_t stream)
{
    (void)in_sizes; (void)n_in; (void)out_size; (void)ws_size;

    const float* x  = (const float*)d_in[0];
    const float* cin[3] = {(const float*)d_in[1], (const float*)d_in[3], (const float*)d_in[5]};
    const float* hin[3] = {(const float*)d_in[2], (const float*)d_in[4], (const float*)d_in[6]};
    const float* Wi[3]  = {(const float*)d_in[7],  (const float*)d_in[10], (const float*)d_in[13]};
    const float* Wh[3]  = {(const float*)d_in[8],  (const float*)d_in[11], (const float*)d_in[14]};
    const float* bb[3]  = {(const float*)d_in[9],  (const float*)d_in[12], (const float*)d_in[15]};
    const float* Wd = (const float*)d_in[16];
    const float* bd = (const float*)d_in[17];
    float* out = (float*)d_out;

    // workspace carve-up (256B aligned)
    char* ws = (char*)d_ws;
    size_t off = 0;
    auto carve = [&](size_t bytes) -> char* {
        off = (off + 255) & ~(size_t)255;
        char* p = ws + off;
        off += bytes;
        return p;
    };
    float*          xW    = (float*)carve((size_t)NROWS * FOURH * 4);  // 128 MB
    unsigned short* xbf   = (unsigned short*)carve((size_t)NROWS * H_ * 2);
    unsigned short* obf[3];
    for (int l = 0; l < 3; l++) obf[l] = (unsigned short*)carve((size_t)NROWS * H_ * 2);
    unsigned short* Wibf[3];
    for (int l = 0; l < 3; l++) Wibf[l] = (unsigned short*)carve((size_t)H_ * FOURH * 2);
    unsigned short* Whbf[3];
    for (int l = 0; l < 3; l++) Whbf[l] = (unsigned short*)carve((size_t)H_ * FOURH * 2);
    unsigned short* Wdbf  = (unsigned short*)carve((size_t)H_ * V_ * 2);
    unsigned short* h_buf = (unsigned short*)carve((size_t)2 * B_ * H_ * 2);
    int*            bar   = (int*)carve(256);

    init_barrier_k<<<1, 64, 0, stream>>>(bar);

    auto cvt = [&](const float* s, unsigned short* d, long n) {
        int blocks = (int)((n + 255) / 256);
        if (blocks > 2048) blocks = 2048;
        f32_to_bf16_k<<<blocks, 256, 0, stream>>>(s, d, n);
    };
    cvt(x, xbf, (long)NROWS * H_);
    for (int l = 0; l < 3; l++) {
        cvt(Wi[l], Wibf[l], (long)H_ * FOURH);
        cvt(Wh[l], Whbf[l], (long)H_ * FOURH);
    }
    cvt(Wd, Wdbf, (long)H_ * V_);

    const unsigned short* layer_in = xbf;
    for (int l = 0; l < 3; l++) {
        // xW = layer_in @ Wi + b   (one big GEMM across all timesteps)
        gemm_bf16_bias_k<<<dim3(FOURH / 128, NROWS / 128), 256, 0, stream>>>(
            layer_in, Wibf[l], bb[l], xW, NROWS, FOURH, H_);
        // sequential scan over T with device-wide barrier per step
        lstm_scan_k<<<NWG_SCAN, 256, 0, stream>>>(
            xW, Whbf[l], cin[l], hin[l], obf[l], h_buf, bar);
        layer_in = obf[l];
    }

    // logits = o3 @ Wd + bd  (N=65, column-guarded)
    gemm_bf16_bias_k<<<dim3(1, NROWS / 128), 256, 0, stream>>>(
        obf[2], Wdbf, bd, out, NROWS, V_, H_);
}
